// Attention_65798898975102
// MI455X (gfx1250) — matrix operand order repaired and checked by reading, not compiled, not run on hardware
//
#include <hip/hip_runtime.h>

typedef __attribute__((ext_vector_type(16))) _Float16 v16h;
typedef __attribute__((ext_vector_type(8)))  _Float16 v8h;
typedef __attribute__((ext_vector_type(8)))  float    v8f;
typedef __attribute__((ext_vector_type(4)))  unsigned int v4u;
typedef __attribute__((ext_vector_type(8)))  int      v8i;
typedef __attribute__((ext_vector_type(4)))  int      v4i;

constexpr int cB  = 4;
constexpr int cL  = 2048;
constexpr int cH  = 1024;
constexpr int cC  = 1024;
constexpr int cD  = 64;
constexpr int cNH = 16;
constexpr float cEPS = 1e-6f;

// ---------------------------------------------------------------------------
// WMMA fragment helpers (layouts per CDNA5 ISA 7.12.2)
// A (16x32 f16): lane m = lane%16; halves 0-7 -> K = g*8..g*8+7,
//                halves 8-15 -> K = 16+g*8 .. 23+g*8   (g = lane/16)
// B (32x16 f16): lane n = lane%16; same K mapping as A (g*8.., 16+g*8..)
// C/D (16x16 f32): lane n = lane%16; element i -> row m = i + 8*g
// ---------------------------------------------------------------------------
__device__ __forceinline__ v16h load_frag_a(const _Float16* rowp, int g) {
    union { v16h v; v8h h[2]; } u;
    u.h[0] = *(const v8h*)(rowp + g * 8);
    u.h[1] = *(const v8h*)(rowp + 16 + g * 8);
    return u.v;
}
__device__ __forceinline__ v16h load_frag_b(const _Float16* colp, int g) {
    union { v16h v; v8h h[2]; } u;
    u.h[0] = *(const v8h*)(colp + g * 8);
    u.h[1] = *(const v8h*)(colp + 16 + g * 8);
    return u.v;
}
__device__ __forceinline__ v8f wmma_f16(v16h a, v16h b, v8f c) {
    return __builtin_amdgcn_wmma_f32_16x16x32_f16(
        /*neg_a=*/false, a, /*neg_b=*/false, b,
        /*c_mod=*/(short)0, c, /*reuse_a=*/false, /*reuse_b=*/false);
}

// ---------------------------------------------------------------------------
// Tensor Data Mover: 2D f16 tile DMA global->LDS with LDS row padding.
// D# layout per CDNA5 ISA 8.3/8.4: group0 = {flags, lds_addr, gaddr_lo,
// gaddr_hi|type=2}; group1 = {dsize/pad, dims, tile dims, dim0 stride}.
// 6-arg builtin form (this toolchain): (g0, g1, g2, g3, g4, cpol).
// Issued by one wave; completion via s_wait_tensorcnt.
// ---------------------------------------------------------------------------
__device__ __forceinline__ void tdm_load_2d_f16(
    const _Float16* gaddr, unsigned lds_byte_off,
    unsigned tile_w, unsigned tile_h, unsigned row_stride_elems,
    unsigned pad_interval_code, unsigned pad_amount_code) {
    unsigned long long ga = (unsigned long long)gaddr;
    v4u g0;
    g0[0] = 1u;                                           // count=1, user D#
    g0[1] = lds_byte_off;                                 // lds_addr
    g0[2] = (unsigned)ga;                                 // global_addr[31:0]
    g0[3] = (unsigned)((ga >> 32) & 0x01FFFFFFu) | 0x80000000u; // [56:32]|type=2
    v8i g1;
    g1[0] = (int)((1u << 16) |                            // data_size = 2 bytes
                  (1u << 20) |                            // pad_enable
                  (pad_interval_code << 22) |
                  (pad_amount_code << 25));
    g1[1] = (int)(tile_w << 16);                          // tensor_dim0 (lo16)
    g1[2] = (int)(tile_h << 16);                          // tensor_dim1 (lo16)
    g1[3] = (int)(tile_w << 16);                          // tile_dim0
    g1[4] = (int)tile_h;                                  // tile_dim1, tile_dim2=0
    g1[5] = (int)row_stride_elems;                        // tensor_dim0_stride lo32
    g1[6] = 0;
    g1[7] = 0;
    v4i gz4 = {0, 0, 0, 0};
    v8i gz8 = {0, 0, 0, 0, 0, 0, 0, 0};
    __builtin_amdgcn_tensor_load_to_lds(g0, g1, gz4, gz4, gz8, 0);
}

__device__ __forceinline__ float group16_max(float v) {
    v = fmaxf(v, __shfl_xor(v, 1, 32));
    v = fmaxf(v, __shfl_xor(v, 2, 32));
    v = fmaxf(v, __shfl_xor(v, 4, 32));
    v = fmaxf(v, __shfl_xor(v, 8, 32));
    return v;
}
__device__ __forceinline__ float group16_sum(float v) {
    v += __shfl_xor(v, 1, 32);
    v += __shfl_xor(v, 2, 32);
    v += __shfl_xor(v, 4, 32);
    v += __shfl_xor(v, 8, 32);
    return v;
}
__device__ __forceinline__ float wave32_sum(float v) {
    v += __shfl_xor(v, 1, 32);
    v += __shfl_xor(v, 2, 32);
    v += __shfl_xor(v, 4, 32);
    v += __shfl_xor(v, 8, 32);
    v += __shfl_xor(v, 16, 32);
    return v;
}

// ---------------------------------------------------------------------------
// 0) fp32 -> f16 convert (weights)
// ---------------------------------------------------------------------------
__global__ void f32_to_f16_kernel(const float* __restrict__ src,
                                  _Float16* __restrict__ dst, int n) {
    int i = blockIdx.x * blockDim.x + threadIdx.x;
    if (i < n) dst[i] = (_Float16)src[i];
}

// ---------------------------------------------------------------------------
// 1) ada = condition @ W_ada + b_ada  -> (B, 2H) fp32 (tiny GEMM, VALU)
// ---------------------------------------------------------------------------
__global__ void ada_kernel(const float* __restrict__ cond,
                           const float* __restrict__ W_ada,
                           const float* __restrict__ b_ada,
                           float* __restrict__ ada) {
    int j = blockIdx.x * blockDim.x + threadIdx.x;     // 0 .. 2H-1
    int b = blockIdx.y;
    if (j >= 2 * cH) return;
    float acc = b_ada[j];
    const float* cp = cond + (size_t)b * cC;
    const float* wp = W_ada + j;
#pragma unroll 4
    for (int c = 0; c < cC; ++c)
        acc += cp[c] * wp[(size_t)c * (2 * cH)];
    ada[(size_t)b * (2 * cH) + j] = acc;
}

// ---------------------------------------------------------------------------
// 2) h = rms_norm(x) * (scale+1) + shift  -> f16   (one block per row)
// ---------------------------------------------------------------------------
__global__ __launch_bounds__(256)
void hnorm_kernel(const float* __restrict__ x, const float* __restrict__ ada,
                  _Float16* __restrict__ h16) {
    int row = blockIdx.x;                  // 0 .. B*L-1
    int b   = row / cL;
    const float* xr = x + (size_t)row * cH;
    float4 v = ((const float4*)xr)[threadIdx.x];
    float ss = v.x * v.x + v.y * v.y + v.z * v.z + v.w * v.w;
    ss = wave32_sum(ss);
    __shared__ float warpsum[8];
    int wid = threadIdx.x >> 5;
    if ((threadIdx.x & 31) == 0) warpsum[wid] = ss;
    __syncthreads();
    float tot = 0.f;
#pragma unroll
    for (int i = 0; i < 8; ++i) tot += warpsum[i];
    float rstd = rsqrtf(tot * (1.0f / cH) + cEPS);

    const float* sh = ada + (size_t)b * (2 * cH);
    int j0 = threadIdx.x * 4;
    float xv[4] = {v.x, v.y, v.z, v.w};
#pragma unroll
    for (int t = 0; t < 4; ++t) {
        int j = j0 + t;
        float y = xv[t] * rstd * (sh[cH + j] + 1.0f) + sh[j];
        h16[(size_t)row * cH + j] = (_Float16)y;
    }
}

// ---------------------------------------------------------------------------
// 3) WMMA GEMM: C(MxN,f32) = A(MxK,f16,row-major) @ B(KxN,f16,row-major)
//    256 threads (8 waves), 128x128 tile, K-step 32.
//    A tile staged by the Tensor Data Mover (TDM) with LDS row padding
//    (32 halves = 16 DWORDs, +4 DWORDs pad -> stride 40); B tile staged
//    (transposed) by the VALU lanes in parallel with the DMA.
// ---------------------------------------------------------------------------
__global__ __launch_bounds__(256)
void gemm_f16f32_kernel(const _Float16* __restrict__ A,
                        const _Float16* __restrict__ Bm,
                        float* __restrict__ Cm, int M, int N, int K) {
    __shared__ __align__(16) _Float16 ldsA[128 * 40];    // row-major, stride 40
    __shared__ __align__(16) _Float16 ldsBT[128 * 40];   // n-major (transposed)
    const int tid  = threadIdx.x;
    const int wave = tid >> 5, lane = tid & 31;
    const int g = lane >> 4, ln = lane & 15;
    const int mh = wave >> 2;          // 0..1: which 64-row half
    const int nq = wave & 3;           // 0..3: which 32-col quarter
    const int rowBase = blockIdx.x * 128;
    const int colBase = blockIdx.y * 128;

    const _Float16* pAtile = A + (size_t)rowBase * K;     // advance 32/iter
    const unsigned ldsA_off = (unsigned)(unsigned long long)&ldsA[0];

    // B = 512 8-half chunks, 2/thread, transposed into n-major LDS
    const int bkr = tid >> 4;                 // 0..15 (and +16)
    const int bnc = (tid & 15) * 8;
    const _Float16* pB0 = Bm + (size_t)bkr * N + colBase + bnc;
    const _Float16* pB1 = pB0 + (size_t)16 * N;
    const size_t bstep = (size_t)32 * N;

    v8f acc[4][2] = {};
    for (int k0 = 0; k0 < K; k0 += 32) {
        // async DMA of the 128x32 A tile into padded LDS (one wave issues)
        if (tid < 32) {
            tdm_load_2d_f16(pAtile, ldsA_off, /*w=*/32, /*h=*/128,
                            /*stride=*/(unsigned)K,
                            /*pad_int=*/3 /*16 DW*/, /*pad_amt=*/3 /*4 DW*/);
        }
        pAtile += 32;
        // meanwhile: stage B tile transposed
        v8h b0 = *(const v8h*)pB0;  pB0 += bstep;
        v8h b1 = *(const v8h*)pB1;  pB1 += bstep;
#pragma unroll
        for (int i = 0; i < 8; ++i) ldsBT[(bnc + i) * 40 + bkr] = b0[i];
#pragma unroll
        for (int i = 0; i < 8; ++i) ldsBT[(bnc + i) * 40 + bkr + 16] = b1[i];
        __builtin_prefetch(pB0, 0, 1);   // next K-step (speculative-safe)
        __builtin_prefetch(pB1, 0, 1);
        if (tid < 32) __builtin_amdgcn_s_wait_tensorcnt(0);
        __syncthreads();

        v16h bf[2];
#pragma unroll
        for (int nt = 0; nt < 2; ++nt)
            bf[nt] = load_frag_b(&ldsBT[(nq * 32 + nt * 16 + ln) * 40], g);
#pragma unroll
        for (int mt = 0; mt < 4; ++mt) {
            v16h af = load_frag_a(&ldsA[(mh * 64 + mt * 16 + ln) * 40], g);
#pragma unroll
            for (int nt = 0; nt < 2; ++nt)
                acc[mt][nt] = wmma_f16(af, bf[nt], acc[mt][nt]);
        }
        __syncthreads();
    }
    // epilogue: incremental pointer walk (rows stride N)
    const int col = colBase + nq * 32 + ln;
#pragma unroll
    for (int mt = 0; mt < 4; ++mt) {
        float* cp = Cm + (size_t)(rowBase + mh * 64 + mt * 16 + 8 * g) * N + col;
#pragma unroll
        for (int i = 0; i < 8; ++i) {
            cp[0]  = acc[mt][0][i];
            cp[16] = acc[mt][1][i];
            cp += N;
        }
    }
}

// ---------------------------------------------------------------------------
// 4) per-head q/k RMSNorm + RoPE, v passthrough. One wave per (b,l,nh);
//    lane owns the RoPE rotation pair (2*lane, 2*lane+1).
//    qkv fp32 (B*L, 3H); outputs f16 in (B,NH,L,D).
// ---------------------------------------------------------------------------
__global__ __launch_bounds__(256)
void qkrope_kernel(const float* __restrict__ qkv, const float* __restrict__ rope,
                   const float* __restrict__ q_scale,
                   const float* __restrict__ k_scale,
                   _Float16* __restrict__ q16, _Float16* __restrict__ k16,
                   _Float16* __restrict__ v16o) {
    int task = blockIdx.x * 8 + (threadIdx.x >> 5);   // (b*L + l)*NH + nh
    int lane = threadIdx.x & 31;
    int nh = task % cNH;
    int bl = task / cNH;
    int l  = bl % cL;
    int b  = bl / cL;
    size_t rowbase = (size_t)bl * (3 * cH);
    size_t outb = ((size_t)(b * cNH + nh) * cL + l) * cD;
    int d0 = lane * 2;

    float2 cs = ((const float2*)(rope + (size_t)l * cD))[lane];
    float2 sn = ((const float2*)(rope + (size_t)cL * cD + (size_t)l * cD))[lane];

    // q
    {
        const float* qp = qkv + rowbase + nh * cD;
        float2 xv = ((const float2*)qp)[lane];
        float ss = wave32_sum(xv.x * xv.x + xv.y * xv.y);
        float rstd = rsqrtf(ss * (1.0f / cD) + cEPS);
        float2 sc = ((const float2*)q_scale)[lane];
        float y0 = xv.x * rstd * sc.x, y1 = xv.y * rstd * sc.y;
        q16[outb + d0]     = (_Float16)(y0 * cs.x - y1 * sn.x);
        q16[outb + d0 + 1] = (_Float16)(y1 * cs.y + y0 * sn.y);
    }
    // k
    {
        const float* kp = qkv + rowbase + cH + nh * cD;
        float2 xv = ((const float2*)kp)[lane];
        float ss = wave32_sum(xv.x * xv.x + xv.y * xv.y);
        float rstd = rsqrtf(ss * (1.0f / cD) + cEPS);
        float2 sc = ((const float2*)k_scale)[lane];
        float y0 = xv.x * rstd * sc.x, y1 = xv.y * rstd * sc.y;
        k16[outb + d0]     = (_Float16)(y0 * cs.x - y1 * sn.x);
        k16[outb + d0 + 1] = (_Float16)(y1 * cs.y + y0 * sn.y);
    }
    // v
    {
        const float* vp = qkv + rowbase + 2 * cH + nh * cD;
        float2 xv = ((const float2*)vp)[lane];
        v16o[outb + d0]     = (_Float16)xv.x;
        v16o[outb + d0 + 1] = (_Float16)xv.y;
    }
}

// ---------------------------------------------------------------------------
// 5) Flash attention. Block = 128 threads (4 waves) handles one
//    (b, head, 64-query tile). K block DMA'd by TDM into padded LDS
//    (64 halves = 32 DWORDs, +4 DWORDs pad -> stride 72); V transposed
//    by VALU lanes in parallel. Q fragments stay in registers. Online
//    softmax on the WMMA C-layout; P reshaped through per-wave LDS.
// ---------------------------------------------------------------------------
__global__ __launch_bounds__(128)
void attn_kernel(const _Float16* __restrict__ q16,
                 const _Float16* __restrict__ k16,
                 const _Float16* __restrict__ v16,
                 _Float16* __restrict__ o16) {
    __shared__ __align__(16) _Float16 ldsK[64 * 72];     // key-major (d contiguous)
    __shared__ __align__(16) _Float16 ldsVT[64 * 72];    // d-major   (key contiguous)
    __shared__ __align__(16) _Float16 ldsP[4][16 * 72];  // per-wave P scratch

    const int tid = threadIdx.x, wave = tid >> 5, lane = tid & 31;
    const int g = lane >> 4, ln = lane & 15;
    const int bh = blockIdx.y;                 // b*NH + nh
    const int qbase = blockIdx.x * 64;
    const _Float16* qhead = q16 + (size_t)bh * cL * cD;
    const _Float16* khead = k16 + (size_t)bh * cL * cD;
    const _Float16* vhead = v16 + (size_t)bh * cL * cD;
    const unsigned ldsK_off = (unsigned)(unsigned long long)&ldsK[0];

    const int qrow = qbase + wave * 16 + ln;
    v16h qf0 = load_frag_a(qhead + (size_t)qrow * cD, g);
    v16h qf1 = load_frag_a(qhead + (size_t)qrow * cD + 32, g);

    // fixed cooperative assignment for the V transpose: 512 chunks, 4/thread
    const int kr0 = tid >> 3;                 // 0..15 (+16/+32/+48)
    const int kdc = (tid & 7) * 8;
    const _Float16* pK = khead;               // TDM source, advance 64 rows/iter
    const _Float16* pV = vhead + (size_t)kr0 * cD + kdc;

    v8f oacc[4] = {};
    float mrow[8], lrow[8];
#pragma unroll
    for (int i = 0; i < 8; ++i) { mrow[i] = -__builtin_inff(); lrow[i] = 0.f; }

    for (int kb = 0; kb < cL; kb += 64) {
        // async DMA of the 64x64 K block into padded LDS (one wave issues)
        if (tid < 32) {
            tdm_load_2d_f16(pK, ldsK_off, /*w=*/64, /*h=*/64, /*stride=*/cD,
                            /*pad_int=*/4 /*32 DW*/, /*pad_amt=*/3 /*4 DW*/);
        }
        pK += (size_t)64 * cD;
        // meanwhile: stage V block transposed
#pragma unroll
        for (int it = 0; it < 4; ++it) {
            int r = kr0 + it * 16;
            v8h tmp = *(const v8h*)(pV + (size_t)it * 16 * cD);
#pragma unroll
            for (int i = 0; i < 8; ++i) ldsVT[(kdc + i) * 72 + r] = tmp[i];
        }
        pV += (size_t)64 * cD;
        __builtin_prefetch(pK, 0, 1);   // next key block (speculative-safe)
        __builtin_prefetch(pV, 0, 1);
        if (tid < 32) __builtin_amdgcn_s_wait_tensorcnt(0);
        __syncthreads();

        // S = (Q @ K^T) / sqrt(D): 4 key n-tiles x 2 k-chunks
        v8f sacc[4] = {};
#pragma unroll
        for (int t = 0; t < 4; ++t) {
            const _Float16* kcol = &ldsK[(t * 16 + ln) * 72];
            sacc[t] = wmma_f16(qf0, load_frag_b(kcol, g), sacc[t]);
            sacc[t] = wmma_f16(qf1, load_frag_b(kcol + 32, g), sacc[t]);
        }

        // online softmax (rows m = i + 8*g, cols across 16-lane group)
        float pf[4][8];
#pragma unroll
        for (int i = 0; i < 8; ++i) {
#pragma unroll
            for (int t = 0; t < 4; ++t) sacc[t][i] *= 0.125f;  // 1/sqrt(64)
            float rmax = fmaxf(fmaxf(sacc[0][i], sacc[1][i]),
                               fmaxf(sacc[2][i], sacc[3][i]));
            rmax = group16_max(rmax);
            float mnew = fmaxf(mrow[i], rmax);
            float corr = __expf(mrow[i] - mnew);
            mrow[i] = mnew;
            float rsum = 0.f;
#pragma unroll
            for (int t = 0; t < 4; ++t) {
                float p = __expf(sacc[t][i] - mnew);
                pf[t][i] = p;
                rsum += p;
            }
            rsum = group16_sum(rsum);
            lrow[i] = lrow[i] * corr + rsum;
#pragma unroll
            for (int t = 0; t < 4; ++t) oacc[t][i] *= corr;
        }

        // reshape P (C-layout) -> A-layout via per-wave LDS scratch
        _Float16* pbuf = ldsP[wave];
#pragma unroll
        for (int t = 0; t < 4; ++t)
#pragma unroll
            for (int i = 0; i < 8; ++i)
                pbuf[(i + 8 * g) * 72 + t * 16 + ln] = (_Float16)pf[t][i];
        asm volatile("s_wait_dscnt 0" ::: "memory");   // same-wave LDS RAW

        v16h pa0 = load_frag_a(&pbuf[ln * 72], g);
        v16h pa1 = load_frag_a(&pbuf[ln * 72 + 32], g);
#pragma unroll
        for (int t = 0; t < 4; ++t) {
            const _Float16* vcol = &ldsVT[(t * 16 + ln) * 72];
            oacc[t] = wmma_f16(pa0, load_frag_b(vcol, g), oacc[t]);
            oacc[t] = wmma_f16(pa1, load_frag_b(vcol + 32, g), oacc[t]);
        }
        __syncthreads();
    }

    // epilogue: o[b, q, nh*D + d] = oacc / l (incremental pointer walk)
    int b = bh / cNH, nh = bh % cNH;
    _Float16* op = o16 + ((size_t)b * cL + qbase + wave * 16 + 8 * g) * cH
                 + nh * cD + ln;
#pragma unroll
    for (int i = 0; i < 8; ++i) {
        float inv = 1.0f / lrow[i];
#pragma unroll
        for (int t = 0; t < 4; ++t)
            op[t * 16] = (_Float16)(oacc[t][i] * inv);
        op += cH;
    }
}

// ---------------------------------------------------------------------------
// launch
// ---------------------------------------------------------------------------
extern "C" void kernel_launch(void* const* d_in, const int* in_sizes, int n_in,
                              void* d_out, int out_size, void* d_ws, size_t ws_size,
                              hipStream_t stream) {
    const float* x       = (const float*)d_in[0];
    const float* cond    = (const float*)d_in[1];
    const float* rope    = (const float*)d_in[2];
    const float* W_ada   = (const float*)d_in[3];
    const float* b_ada   = (const float*)d_in[4];
    const float* W_qkv   = (const float*)d_in[5];
    const float* W_out   = (const float*)d_in[6];
    const float* q_scale = (const float*)d_in[7];
    const float* k_scale = (const float*)d_in[8];
    float* out = (float*)d_out;

    char* ws = (char*)d_ws;
    size_t off = 0;
    auto alloc = [&](size_t bytes) -> void* {
        void* p = ws + off;
        off += (bytes + 255) & ~(size_t)255;
        return p;
    };
    const size_t rows = (size_t)cB * cL;                     // 8192
    float*    ada    = (float*)   alloc((size_t)cB * 2 * cH * sizeof(float));
    _Float16* h16    = (_Float16*)alloc(rows * cH * 2);
    _Float16* wqkv16 = (_Float16*)alloc((size_t)cH * 3 * cH * 2);
    _Float16* wout16 = (_Float16*)alloc((size_t)cH * cH * 2);
    float*    qkv32  = (float*)   alloc(rows * 3 * cH * sizeof(float));
    _Float16* q16    = (_Float16*)alloc(rows * cH * 2);
    _Float16* k16    = (_Float16*)alloc(rows * cH * 2);
    _Float16* v16    = (_Float16*)alloc(rows * cH * 2);
    _Float16* o16    = (_Float16*)alloc(rows * cH * 2);
    (void)ws_size; (void)in_sizes; (void)n_in; (void)out_size;

    // weight converts
    {
        int n = cH * 3 * cH;
        f32_to_f16_kernel<<<(n + 255) / 256, 256, 0, stream>>>(W_qkv, wqkv16, n);
        n = cH * cH;
        f32_to_f16_kernel<<<(n + 255) / 256, 256, 0, stream>>>(W_out, wout16, n);
    }
    // adaLN
    ada_kernel<<<dim3((2 * cH) / 256, cB), 256, 0, stream>>>(cond, W_ada, b_ada, ada);
    // h = modulated rmsnorm(x) -> f16
    hnorm_kernel<<<(int)rows, 256, 0, stream>>>(x, ada, h16);
    // qkv = h @ W_qkv  (8192 x 3072, K=1024), 128x128 tiles
    gemm_f16f32_kernel<<<dim3((int)rows / 128, (3 * cH) / 128), 256, 0, stream>>>(
        h16, wqkv16, qkv32, (int)rows, 3 * cH, cH);
    // q/k norm + rope, v convert -> (B,NH,L,D) f16
    qkrope_kernel<<<(int)(rows * cNH / 8), 256, 0, stream>>>(
        qkv32, rope, q_scale, k_scale, q16, k16, v16);
    // flash attention -> o16 (B*L, H)
    attn_kernel<<<dim3(cL / 64, cB * cNH), 128, 0, stream>>>(q16, k16, v16, o16);
    // out = o @ W_out  (8192 x 1024, K=1024) -> fp32
    gemm_f16f32_kernel<<<dim3((int)rows / 128, cH / 128), 256, 0, stream>>>(
        o16, wout16, out, (int)rows, cH, cH);
}
